// Head_21414706937875
// MI455X (gfx1250) — matrix operand order repaired and checked by reading, not compiled, not run on hardware
//
#include <hip/hip_runtime.h>

typedef __attribute__((ext_vector_type(16))) _Float16 v16h;
typedef __attribute__((ext_vector_type(8)))  _Float16 v8h;
typedef __attribute__((ext_vector_type(4)))  _Float16 v4h;
typedef __attribute__((ext_vector_type(4)))  float    v4f;
typedef __attribute__((ext_vector_type(8)))  float    v8f;

#define N_EMBD 1024
#define HEAD   64
#define WIN    256
#define BATCH  4
#define SEQ    4096
#define ROWS   (BATCH * SEQ)        // 16384
#define NEG_BIG (-1.0e30f)

static __device__ inline v8f zero8() {
    v8f z;
#pragma unroll
    for (int i = 0; i < 8; ++i) z[i] = 0.0f;
    return z;
}

static __device__ inline v16h cat8(v8h lo, v8h hi) {
    return __builtin_shufflevector(lo, hi, 0, 1, 2, 3, 4, 5, 6, 7,
                                           8, 9, 10, 11, 12, 13, 14, 15);
}

// ---------------------------------------------------------------------------
// Kernel 1: fused q/k/v projection.
//   x [16384,1024] fp32 -> q,k [16384,64] f16 (row-major) and vT [4,64,4096] f16.
// 8 waves per WG, each wave owns a 16-row tile; WG covers 128 rows.
// LDS: W chunk transposed (col-major -> B fragments contiguous per lane),
//      x chunk row-major (A fragments = one contiguous 32B run per lane).
// Per W matrix: batch-load 4 B fragments, then 4 back-to-back WMMAs.
// ---------------------------------------------------------------------------
__global__ __launch_bounds__(256) void proj_kernel(
    const float* __restrict__ x,
    const float* __restrict__ Wq,
    const float* __restrict__ Wk,
    const float* __restrict__ Wv,
    _Float16* __restrict__ qh,
    _Float16* __restrict__ kh,
    _Float16* __restrict__ vT)
{
    __shared__ _Float16 wldsT[192 * 32];  // 12 KB: [col 0..191][K 0..31], cols=[Wq|Wk|Wv]
    __shared__ _Float16 xlds[128 * 32];   // 8 KB : [row][K]

    const int t    = threadIdx.x;
    const int wave = t >> 5;
    const int lane = t & 31;
    const int lh   = (lane >> 4) & 1;     // wave half: 0 = lanes 0-15, 1 = 16-31
    const int l16  = lane & 15;

    const int wgRow0 = blockIdx.x * 128;
    const int myRow0 = wgRow0 + wave * 16;

    v8f acc[12];                          // [m*4 + n]
#pragma unroll
    for (int i = 0; i < 12; ++i) acc[i] = zero8();

    for (int k0 = 0; k0 < N_EMBD; k0 += 32) {
        // ---- stage W slice transposed: float4 global reads, b16 LDS scatter ----
#pragma unroll
        for (int i = 0; i < 2; ++i) {                 // 512 float4 per matrix
            const int idx4 = t + i * 256;
            const int r  = idx4 >> 4;                 // K row 0..31
            const int c0 = (idx4 & 15) << 2;          // col 0..60
            const v4f wq = *(const v4f*)&Wq[(k0 + r) * HEAD + c0];
            const v4f wk = *(const v4f*)&Wk[(k0 + r) * HEAD + c0];
            const v4f wv = *(const v4f*)&Wv[(k0 + r) * HEAD + c0];
#pragma unroll
            for (int j = 0; j < 4; ++j) {
                wldsT[(c0 + j) * 32 + r]       = (_Float16)wq[j];
                wldsT[(64 + c0 + j) * 32 + r]  = (_Float16)wk[j];
                wldsT[(128 + c0 + j) * 32 + r] = (_Float16)wv[j];
            }
        }
        // ---- stage x chunk: 128 rows x 32 cols, float4 -> packed 4xf16 ----
#pragma unroll
        for (int i = 0; i < 4; ++i) {                 // 1024 float4 total
            const int idx4 = t + i * 256;
            const int r  = idx4 >> 3;                 // row 0..127 (8 float4/row)
            const int c4 = (idx4 & 7) << 2;
            const v4f xv = *(const v4f*)&x[(size_t)(wgRow0 + r) * N_EMBD + k0 + c4];
            v4h hx;
#pragma unroll
            for (int j = 0; j < 4; ++j) hx[j] = (_Float16)xv[j];
            *(v4h*)&xlds[r * 32 + c4] = hx;
        }
        if (k0 + 32 < N_EMBD) {
            __builtin_prefetch(&x[(size_t)(wgRow0 + (t >> 1)) * N_EMBD + k0 + 32], 0, 3);
        }
        __syncthreads();

        // ---- A fragment: one contiguous 32B run per lane ----
        const _Float16* xrow = &xlds[(wave * 16 + l16) * 32];
        const v16h a = *(const v16h*)(xrow + lh * 16);

        // ---- per W matrix: load 4 B fragments, then 4 back-to-back WMMAs ----
#pragma unroll
        for (int m = 0; m < 3; ++m) {
            v16h bf[4];
#pragma unroll
            for (int n = 0; n < 4; ++n)
                bf[n] = *(const v16h*)&wldsT[(m * 64 + n * 16 + l16) * 32 + lh * 16];
#pragma unroll
            for (int n = 0; n < 4; ++n)
                acc[m * 4 + n] = __builtin_amdgcn_wmma_f32_16x16x32_f16(
                    false, a, false, bf[n], (short)0, acc[m * 4 + n], false, false);
        }
        __syncthreads();
    }

    // ---- store q/k row-major f16; v transposed [b][c][t] f16 ----
    const int bidx    = myRow0 >> 12;          // row / 4096
    const int tlocal0 = myRow0 & 4095;
#pragma unroll
    for (int n = 0; n < 4; ++n) {
#pragma unroll
        for (int r = 0; r < 8; ++r) {
            const int  rowoff = r + 8 * lh;
            const int  col    = n * 16 + l16;
            const size_t idx  = (size_t)(myRow0 + rowoff) * HEAD + col;
            qh[idx] = (_Float16)acc[0 * 4 + n][r];
            kh[idx] = (_Float16)acc[1 * 4 + n][r];
            vT[((size_t)bidx * HEAD + col) * SEQ + tlocal0 + rowoff] = (_Float16)acc[2 * 4 + n][r];
        }
    }
}

// ---------------------------------------------------------------------------
// Kernel 2: sliding-window causal attention, flash-style online softmax.
// One wave per 16-query tile; 1024 tiles; 8 waves/WG -> 128 WGs.
// All fragments are contiguous vector loads; P transposed via 512B per-wave
// LDS bounce; 4 PV WMMAs issued back-to-back after batched bv loads.
// ---------------------------------------------------------------------------
__global__ __launch_bounds__(256) void attn_kernel(
    const _Float16* __restrict__ qh,
    const _Float16* __restrict__ kh,
    const _Float16* __restrict__ vT,
    float* __restrict__ out)
{
    __shared__ _Float16 pbuf[8][16 * 16];  // per-wave P bounce (512B each)

    const int t    = threadIdx.x;
    const int wave = t >> 5;
    const int lane = t & 31;
    const int lh   = (lane >> 4) & 1;
    const int l16  = lane & 15;

    const int tile = blockIdx.x * 8 + wave;   // 0..1023
    const int b    = tile >> 8;               // 256 tiles per batch
    const int qt   = tile & 255;
    const size_t rbase = (size_t)b * SEQ;
    const int qrow0 = qt * 16;

    // ---- Q fragments: one 32B run per lane each ----
    const _Float16* qrow = qh + (rbase + qrow0 + l16) * HEAD;
    const v16h aq0 = *(const v16h*)(qrow + lh * 16);
    const v16h aq1 = *(const v16h*)(qrow + 32 + lh * 16);

    v8f o[4];
#pragma unroll
    for (int n = 0; n < 4; ++n) o[n] = zero8();

    float m_i[8], l_i[8];
#pragma unroll
    for (int r = 0; r < 8; ++r) { m_i[r] = NEG_BIG; l_i[r] = 0.0f; }

    int jt0 = qt - 16; if (jt0 < 0) jt0 = 0;

    for (int jt = jt0; jt <= qt; ++jt) {
        const int krow0 = jt * 16;

        // ---- K^T fragments: one 32B run per lane each ----
        const _Float16* krow = kh + (rbase + krow0 + l16) * HEAD;
        const v16h bk0 = *(const v16h*)(krow + lh * 16);
        const v16h bk1 = *(const v16h*)(krow + 32 + lh * 16);

        // ---- V fragments for this key tile: batch the 4 loads up front ----
        v16h bvf[4];
#pragma unroll
        for (int n = 0; n < 4; ++n)
            bvf[n] = *(const v16h*)&vT[((size_t)b * HEAD + n * 16 + l16) * SEQ + krow0];

        v8f s = zero8();
        s = __builtin_amdgcn_wmma_f32_16x16x32_f16(false, aq0, false, bk0, (short)0, s, false, false);
        s = __builtin_amdgcn_wmma_f32_16x16x32_f16(false, aq1, false, bk1, (short)0, s, false, false);

        // ---- mask + scale + online softmax (row = r + 8*lh, col = l16) ----
#pragma unroll
        for (int r = 0; r < 8; ++r) {
            const int qrowi = qrow0 + r + 8 * lh;
            const int krowi = krow0 + l16;
            const int d = qrowi - krowi;
            float sv = (d >= 0 && d < WIN) ? s[r] * 0.125f : NEG_BIG;

            float mx = sv;
#pragma unroll
            for (int off = 8; off; off >>= 1)
                mx = fmaxf(mx, __shfl_xor(mx, off, 32));   // stays in 16-lane half

            const float mn    = fmaxf(m_i[r], mx);
            const float alpha = __expf(m_i[r] - mn);
            const float p     = __expf(sv - mn);

            float ps = p;
#pragma unroll
            for (int off = 8; off; off >>= 1)
                ps += __shfl_xor(ps, off, 32);

            l_i[r] = l_i[r] * alpha + ps;
            m_i[r] = mn;
#pragma unroll
            for (int n = 0; n < 4; ++n) o[n][r] *= alpha;

            pbuf[wave][(r + 8 * lh) * 16 + l16] = (_Float16)p;
        }

        // same-wave LDS RAW: wait for the 8 ds_stores before re-reading
        asm volatile("s_wait_dscnt 0" ::: "memory");

        // ---- A = P (16x16, K padded to 32 with zeros in A only) ----
        v8h zpad;
#pragma unroll
        for (int e = 0; e < 8; ++e) zpad[e] = (_Float16)0.0f;
        const v16h ap = lh ? cat8(zpad, zpad)
                           : cat8(*(const v8h*)&pbuf[wave][l16 * 16], *(const v8h*)&pbuf[wave][l16 * 16 + 8]);

        // ---- O += P x V : 4 back-to-back WMMAs.
        //      A's K>=16 elements are zero, so B's upper-K half needs no mask. ----
#pragma unroll
        for (int n = 0; n < 4; ++n)
            o[n] = __builtin_amdgcn_wmma_f32_16x16x32_f16(
                false, ap, false, bvf[n], (short)0, o[n], false, false);
    }

    // ---- normalize and store fp32 output ----
#pragma unroll
    for (int n = 0; n < 4; ++n) {
#pragma unroll
        for (int r = 0; r < 8; ++r) {
            const size_t row = rbase + qrow0 + r + 8 * lh;
            out[row * HEAD + n * 16 + l16] = o[n][r] / l_i[r];
        }
    }
}

// ---------------------------------------------------------------------------
extern "C" void kernel_launch(void* const* d_in, const int* in_sizes, int n_in,
                              void* d_out, int out_size, void* d_ws, size_t ws_size,
                              hipStream_t stream) {
    const float* x  = (const float*)d_in[0];
    const float* Wk = (const float*)d_in[1];
    const float* Wq = (const float*)d_in[2];
    const float* Wv = (const float*)d_in[3];

    _Float16* qh = (_Float16*)d_ws;                     // [16384, 64] f16
    _Float16* kh = qh + (size_t)ROWS * HEAD;            // [16384, 64] f16
    _Float16* vT = kh + (size_t)ROWS * HEAD;            // [4, 64, 4096] f16

    proj_kernel<<<ROWS / 128, 256, 0, stream>>>(x, Wq, Wk, Wv, qh, kh, vT);
    attn_kernel<<<(ROWS / 16) / 8, 256, 0, stream>>>(qh, kh, vT, (float*)d_out);
}
